// MLA_47021301956804
// MI455X (gfx1250) — matrix operand order repaired and checked by reading, not compiled, not run on hardware
//
#include <hip/hip_runtime.h>
#include <hip/hip_bf16.h>

// ---------------------------------------------------------------------------
// MLA forward for MI455X (gfx1250, wave32, WMMA).
// Compute-bound (~210 GFLOP vs ~0.4GB HBM) -> all matmuls through
// v_wmma_f32_16x16x32_f16 (f16 operands, f32 accumulate).
// Data movement: one-time f32->f16 conversion, then GLOBAL_LOAD_ASYNC_TO_LDS
// (ASYNCcnt) staging with double-buffered LDS tiles in the GEMM.
// ---------------------------------------------------------------------------

typedef __attribute__((ext_vector_type(16))) _Float16 v16h;
typedef __attribute__((ext_vector_type(8)))  float    v8f;
typedef __attribute__((ext_vector_type(4)))  int      i4;

constexpr int B_   = 2;
constexpr int S_   = 2048;
constexpr int DIM  = 2048;
constexpr int NH   = 16;
constexpr int HD   = 128;
constexpr int HDR  = 64;
constexpr int HDK  = HD + HDR;      // 192 per-head q/k dim
constexpr int DCKV = 512;
constexpr int DCQ  = 1536;
constexpr int MTOK = B_ * S_;       // 4096 tokens

#define AS1 __attribute__((address_space(1)))
#define AS3 __attribute__((address_space(3)))

#if defined(__gfx1250__) && __has_builtin(__builtin_amdgcn_global_load_async_to_lds_b128)
#define HAS_ASYNC_LDS 1
#else
#define HAS_ASYNC_LDS 0
#endif

// ---------------- helpers ----------------

__device__ __forceinline__ unsigned int pack2_f16(float a, float b) {
    union { _Float16 h[2]; unsigned int u; } u;
    u.h[0] = (_Float16)a; u.h[1] = (_Float16)b;
    return u.u;
}

// 16B global -> LDS copy; async (ASYNCcnt) when available, else load+ds_store.
__device__ __forceinline__ void cp_b128_g2l(const _Float16* g, _Float16* l) {
#if HAS_ASYNC_LDS
    __builtin_amdgcn_global_load_async_to_lds_b128(
        (AS1 i4*)(unsigned long long)(const void*)g,
        (AS3 i4*)(unsigned int)(unsigned long long)(void*)l,
        0, 0);
#else
    *(uint4*)l = *(const uint4*)g;
#endif
}

__device__ __forceinline__ void wait_async0() {
#if HAS_ASYNC_LDS
#if __has_builtin(__builtin_amdgcn_s_wait_asynccnt)
    __builtin_amdgcn_s_wait_asynccnt(0);
#else
    asm volatile("s_wait_asynccnt 0x0" ::: "memory");
#endif
#endif
}

// A-fragment (16x32 f16): lanes 0-15 K in {0..7,16..23}, lanes 16-31 K in {8..15,24..31}
__device__ __forceinline__ v16h load_a_frag(const _Float16* p_row_base, int ld, int lane) {
    const int r  = lane & 15;
    const int kb = (lane >> 4) * 8;
    const _Float16* p = p_row_base + r * ld;
    v16h a;
#pragma unroll
    for (int j = 0; j < 8; ++j) a[j]     = p[kb + j];
#pragma unroll
    for (int j = 0; j < 8; ++j) a[8 + j] = p[kb + 16 + j];
    return a;
}

// B-fragment (32x16 f16): lane = column, lanes 0-15 K in {0..7,16..23}, lanes 16-31 K in {8..15,24..31}.
__device__ __forceinline__ v16h load_b_frag(const _Float16* p_col_base, int ld, int lane) {
    const int n  = lane & 15;
    const int kb = (lane >> 4) * 8;
    const _Float16* p = p_col_base + n * ld + kb;
    v16h b;
#pragma unroll
    for (int j = 0; j < 8; ++j) b[j] = p[j];
#pragma unroll
    for (int j = 0; j < 8; ++j) b[8 + j] = p[16 + j];
    return b;
}

// ---------------- f32 -> f16 conversion (one-time, ~4us of HBM traffic) -------

__global__ __launch_bounds__(256) void mla_cvt_h(const float* __restrict__ src,
                                                 _Float16* __restrict__ dst, int n) {
    const int i = (blockIdx.x * blockDim.x + threadIdx.x) * 8;
    if (i >= n) return;
    float4 a = *(const float4*)(src + i);
    float4 b = *(const float4*)(src + i + 4);
    uint4 pk;
    pk.x = pack2_f16(a.x, a.y); pk.y = pack2_f16(a.z, a.w);
    pk.z = pack2_f16(b.x, b.y); pk.w = pack2_f16(b.z, b.w);
    *(uint4*)(dst + i) = pk;
}

// ---------------- generic GEMM: Y[M,N] = A[M,K] * W[N,K]^T + bias ----------------
// f16 inputs. 256 threads = 8 waves; block tile 64x64; wave tile 16x32.
// Double-buffered LDS, async global->LDS staging, one barrier per K-step.
// f16 output column remap (powers of two): c2 = coloff + (c>>gshift)*gstride + (c & (2^gshift-1)).

constexpr int GPAD = 40;  // LDS row stride in halves (80B, 16B aligned, bank-skewed)

__global__ __launch_bounds__(256) void mla_gemm_h(
    const _Float16* __restrict__ A, const _Float16* __restrict__ W,
    const float* __restrict__ bias, int M, int N, int K,
    float* __restrict__ out32, int ldo32,
    _Float16* __restrict__ out16, int ldo16, int gshift, int gstride, int coloff)
{
    __shared__ _Float16 As[2][64 * GPAD];
    __shared__ _Float16 Ws[2][64 * GPAD];

    const int tid  = threadIdx.x;
    const int lane = tid & 31;
    const int wave = tid >> 5;
    const int wm   = wave & 3;   // wave row (4)
    const int wn   = wave >> 2;  // wave col (2)
    const int blockM = blockIdx.y * 64;
    const int blockN = blockIdx.x * 64;

    const int srow = tid >> 2;        // 0..63
    const int scol = (tid & 3) * 8;   // halves: 0,8,16,24 (16B chunks)

    const _Float16* arow = A + (size_t)(blockM + srow) * K + scol;
    const _Float16* wrow = W + (size_t)(blockN + srow) * K + scol;
    _Float16* adst0 = &As[0][srow * GPAD + scol];
    _Float16* wdst0 = &Ws[0][srow * GPAD + scol];
    _Float16* adst1 = &As[1][srow * GPAD + scol];
    _Float16* wdst1 = &Ws[1][srow * GPAD + scol];

    v8f acc0 = {};
    v8f acc1 = {};

    // prefetch tile 0
    cp_b128_g2l(arow, adst0);
    cp_b128_g2l(wrow, wdst0);

    for (int k0 = 0; k0 < K; k0 += 32) {
        const int buf = (k0 >> 5) & 1;
        wait_async0();
        __syncthreads();                       // tile `buf` ready; previous reads of buf^1 done
        if (k0 + 32 < K) {                     // issue next tile into the other buffer
            cp_b128_g2l(arow + k0 + 32, buf ? adst0 : adst1);
            cp_b128_g2l(wrow + k0 + 32, buf ? wdst0 : wdst1);
        }
        v16h af = load_a_frag(&As[buf][(wm * 16) * GPAD], GPAD, lane);
        v16h b0 = load_b_frag(&Ws[buf][(wn * 32 + 0)  * GPAD], GPAD, lane);
        v16h b1 = load_b_frag(&Ws[buf][(wn * 32 + 16) * GPAD], GPAD, lane);
        acc0 = __builtin_amdgcn_wmma_f32_16x16x32_f16(false, af, false, b0, (short)0, acc0, false, false);
        acc1 = __builtin_amdgcn_wmma_f32_16x16x32_f16(false, af, false, b1, (short)0, acc1, false, false);
    }

    // epilogue: C layout -> lane = column, VGPR r = row r (lanes 0-15) / r+8 (lanes 16-31)
    const int colL   = lane & 15;
    const int rowAdd = (lane >> 4) * 8;
#pragma unroll
    for (int sub = 0; sub < 2; ++sub) {
        v8f acc = sub ? acc1 : acc0;
        const int colg = blockN + wn * 32 + sub * 16 + colL;
        const float bv = bias ? bias[colg] : 0.0f;
        int c2 = 0;
        if (out16) c2 = coloff + ((colg >> gshift) * gstride) + (colg & ((1 << gshift) - 1));
#pragma unroll
        for (int r = 0; r < 8; ++r) {
            const int rowg = blockM + wm * 16 + rowAdd + r;
            const float v = acc[r] + bv;
            if (out32) out32[(size_t)rowg * ldo32 + colg] = v;
            if (out16) out16[(size_t)rowg * ldo16 + c2]   = (_Float16)v;
        }
    }
}

// ---------------- RoPE ----------------

__global__ void mla_rope_q(const float* __restrict__ cqr,
                           const float* __restrict__ cosb, const float* __restrict__ sinb,
                           _Float16* __restrict__ q16)
{
    const int idx = blockIdx.x * blockDim.x + threadIdx.x;   // over B*S*NH*HDR
    if (idx >= B_ * S_ * NH * HDR) return;
    const int d = idx & (HDR - 1);
    const int h = (idx >> 6) & (NH - 1);
    const int t = idx >> 10;             // b*S + s
    const int s = t & (S_ - 1);
    const float* row = cqr + (size_t)t * (NH * HDR) + h * HDR;
    const float x   = row[d];
    const float rot = (d < HDR / 2) ? -row[d + HDR / 2] : row[d - HDR / 2];
    const float v   = x * cosb[s * HDR + d] + rot * sinb[s * HDR + d];
    q16[(size_t)t * (NH * HDK) + h * HDK + HD + d] = (_Float16)v;
}

__global__ void mla_rope_k(const float* __restrict__ krl,
                           const float* __restrict__ cosb, const float* __restrict__ sinb,
                           _Float16* __restrict__ k16)
{
    const int idx = blockIdx.x * blockDim.x + threadIdx.x;   // over B*S*HDR
    if (idx >= B_ * S_ * HDR) return;
    const int d = idx & (HDR - 1);
    const int t = idx >> 6;              // b*S + s
    const int s = t & (S_ - 1);
    const float* row = krl + (size_t)t * HDR;
    const float x   = row[d];
    const float rot = (d < HDR / 2) ? -row[d + HDR / 2] : row[d - HDR / 2];
    const _Float16 v = (_Float16)(x * cosb[s * HDR + d] + rot * sinb[s * HDR + d]);
    _Float16* dst = k16 + (size_t)t * (NH * HDK) + HD + d;
#pragma unroll
    for (int hh = 0; hh < NH; ++hh) dst[hh * HDK] = v;       // broadcast over heads
}

// ---------------- flash attention (causal) ----------------
// Block = (b, h, 64 query rows); 4 waves x 16 rows. 32-key chunks.
// 12 WMMAs (QK^T over 192 feat) + 8 WMMAs (P@V over 128 dims) per chunk.

constexpr int BQ = 64;
constexpr int CH = 32;
constexpr int KP = HDK + 8;   // 200 halves, 400B (16B aligned)
constexpr int VP = CH + 8;    // 40 halves
constexpr int PP = 40;

__global__ __launch_bounds__(128) void mla_attn(
    const _Float16* __restrict__ q16, const _Float16* __restrict__ k16,
    const _Float16* __restrict__ v16, _Float16* __restrict__ y16)
{
    __shared__ _Float16 Kt[CH * KP];          // key-major 32 x 192 (+pad)
    __shared__ _Float16 Vt[HD * VP];          // dim-major 128 x 32 (+pad), i.e. V^T
    __shared__ _Float16 Pl[4 * 16 * PP];      // per-wave P tiles 16x32

    const int tid  = threadIdx.x;
    const int lane = tid & 31;
    const int wave = tid >> 5;
    const int q0 = blockIdx.x * BQ;
    const int h  = blockIdx.y;
    const int b  = blockIdx.z;
    const int qw = q0 + wave * 16;

    const int qk_row = NH * HDK;              // 3072
    const _Float16* qbase = q16 + (size_t)(b * S_) * qk_row + h * HDK;

    // Q fragments for this wave's 16 rows: 6 x (16x32)
    v16h Qf[6];
    {
        const int r  = qw + (lane & 15);
        const int kb = (lane >> 4) * 8;
        const _Float16* qrow = qbase + (size_t)r * qk_row;
#pragma unroll
        for (int fc = 0; fc < 6; ++fc) {
            const _Float16* p = qrow + fc * 32;
#pragma unroll
            for (int j = 0; j < 8; ++j) Qf[fc][j]     = p[kb + j];
#pragma unroll
            for (int j = 0; j < 8; ++j) Qf[fc][8 + j] = p[kb + 16 + j];
        }
    }

    v8f O[8];
    v8f vzero = {};
#pragma unroll
    for (int i = 0; i < 8; ++i) O[i] = vzero;
    float mrow[8], lrow[8];
#pragma unroll
    for (int r = 0; r < 8; ++r) { mrow[r] = -1e30f; lrow[r] = 0.0f; }

    const float scale = rsqrtf((float)HDK);
    const int nchunk = (q0 + BQ) / CH;

    for (int c = 0; c < nchunk; ++c) {
        const int kbase = c * CH;
        // stage K chunk (32 keys x 192 halves) via async copies: 6 x 16B per thread
        {
            const int key = tid >> 2;
            const _Float16* src = k16 + (size_t)(b * S_ + kbase + key) * qk_row + h * HDK;
            _Float16* dst = &Kt[key * KP];
#pragma unroll
            for (int i = 0; i < 6; ++i) {
                const int seg = ((tid & 3) + 4 * i) * 8;   // halves
                cp_b128_g2l(src + seg, dst + seg);
            }
        }
        // stage V chunk transposed -> Vt[dim][key] (transpose: scalar ds stores)
        {
            const int key   = tid >> 2;
            const int dpart = (tid & 3) * 32;
            const _Float16* src = v16 + (size_t)(b * S_ + kbase + key) * (NH * HD) + h * HD + dpart;
#pragma unroll
            for (int i = 0; i < 32; ++i)
                Vt[(dpart + i) * VP + key] = src[i];
        }
        wait_async0();
        __syncthreads();

        // scores: S(16x32) = Q(16x192) * K_chunk^T
        v8f s0 = vzero, s1 = vzero;
#pragma unroll
        for (int fc = 0; fc < 6; ++fc) {
            v16h b0 = load_b_frag(&Kt[0  * KP + fc * 32], KP, lane);
            v16h b1 = load_b_frag(&Kt[16 * KP + fc * 32], KP, lane);
            s0 = __builtin_amdgcn_wmma_f32_16x16x32_f16(false, Qf[fc], false, b0, (short)0, s0, false, false);
            s1 = __builtin_amdgcn_wmma_f32_16x16x32_f16(false, Qf[fc], false, b1, (short)0, s1, false, false);
        }

        // online softmax (row stats per lane-half, 8 rows each)
        const int colL   = lane & 15;
        const int rowAdd = (lane >> 4) * 8;
        float p0v[8], p1v[8], fac[8];
#pragma unroll
        for (int r = 0; r < 8; ++r) {
            const int qrow = qw + rowAdd + r;
            float v0 = s0[r] * scale; if (kbase + colL      > qrow) v0 = -1e30f;
            float v1 = s1[r] * scale; if (kbase + 16 + colL > qrow) v1 = -1e30f;
            float t = fmaxf(v0, v1);
            t = fmaxf(t, __shfl_xor(t, 1, 32));
            t = fmaxf(t, __shfl_xor(t, 2, 32));
            t = fmaxf(t, __shfl_xor(t, 4, 32));
            t = fmaxf(t, __shfl_xor(t, 8, 32));
            const float mnew = fmaxf(mrow[r], t);
            const float f = __expf(mrow[r] - mnew);
            mrow[r] = mnew; fac[r] = f;
            const float e0 = __expf(v0 - mnew);
            const float e1 = __expf(v1 - mnew);
            p0v[r] = e0; p1v[r] = e1;
            float rs = e0 + e1;
            rs += __shfl_xor(rs, 1, 32);
            rs += __shfl_xor(rs, 2, 32);
            rs += __shfl_xor(rs, 4, 32);
            rs += __shfl_xor(rs, 8, 32);
            lrow[r] = lrow[r] * f + rs;
        }
#pragma unroll
        for (int dt = 0; dt < 8; ++dt)
#pragma unroll
            for (int r = 0; r < 8; ++r) O[dt][r] *= fac[r];

        // P: C-layout -> LDS -> A-fragment (per-wave private tile, same-wave DS order)
        _Float16* pw = &Pl[wave * 16 * PP];
#pragma unroll
        for (int r = 0; r < 8; ++r) {
            pw[(rowAdd + r) * PP + colL]      = (_Float16)p0v[r];
            pw[(rowAdd + r) * PP + 16 + colL] = (_Float16)p1v[r];
        }
        v16h Pf = load_a_frag(pw, PP, lane);

        // O(16x128) += P(16x32) * V(32x128)
#pragma unroll
        for (int dt = 0; dt < 8; ++dt) {
            v16h bv = load_b_frag(&Vt[(dt * 16) * VP], VP, lane);
            O[dt] = __builtin_amdgcn_wmma_f32_16x16x32_f16(false, Pf, false, bv, (short)0, O[dt], false, false);
        }
        __syncthreads();
    }

    // epilogue: y = O / l -> (B,S,NH*HD) f16 (feeds final GEMM directly)
    {
        const int colL   = lane & 15;
        const int rowAdd = (lane >> 4) * 8;
#pragma unroll
        for (int r = 0; r < 8; ++r) {
            const int qrow = qw + rowAdd + r;
            const float inv = 1.0f / lrow[r];
            _Float16* dst = y16 + (size_t)(b * S_ + qrow) * (NH * HD) + h * HD;
#pragma unroll
            for (int dt = 0; dt < 8; ++dt)
                dst[dt * 16 + colL] = (_Float16)(O[dt][r] * inv);
        }
    }
}

// ---------------- launcher ----------------

static inline size_t align256(size_t x) { return (x + 255) & ~(size_t)255; }

extern "C" void kernel_launch(void* const* d_in, const int* in_sizes, int n_in,
                              void* d_out, int out_size, void* d_ws, size_t ws_size,
                              hipStream_t stream) {
    const float* x     = (const float*)d_in[0];
    const float* fcos  = (const float*)d_in[1];
    const float* fsin  = (const float*)d_in[2];
    const float* W_dq  = (const float*)d_in[3];
    const float* b_dq  = (const float*)d_in[4];
    const float* W_uq  = (const float*)d_in[5];
    const float* b_uq  = (const float*)d_in[6];
    const float* W_qr  = (const float*)d_in[7];
    const float* b_qr  = (const float*)d_in[8];
    const float* W_dkv = (const float*)d_in[9];
    const float* b_dkv = (const float*)d_in[10];
    const float* W_uk  = (const float*)d_in[11];
    const float* b_uk  = (const float*)d_in[12];
    const float* W_uv  = (const float*)d_in[13];
    const float* b_uv  = (const float*)d_in[14];
    const float* W_kr  = (const float*)d_in[15];
    const float* b_kr  = (const float*)d_in[16];
    const float* W_out = (const float*)d_in[17];
    const float* b_out = (const float*)d_in[18];

    // ---- workspace carve-out ----
    char* ws = (char*)d_ws;
    size_t off = 0;
    auto takeH = [&](size_t n) { _Float16* p = (_Float16*)(ws + off); off = align256(off + n * 2); return p; };
    auto takeF = [&](size_t n) { float* p = (float*)(ws + off); off = align256(off + n * 4); return p; };

    _Float16* x16    = takeH((size_t)MTOK * DIM);
    _Float16* wdq16  = takeH((size_t)DCQ * DIM);
    _Float16* wuq16  = takeH((size_t)NH * HD * DCQ);
    _Float16* wqr16  = takeH((size_t)NH * HDR * DCQ);
    _Float16* wdkv16 = takeH((size_t)DCKV * DIM);
    _Float16* wuk16  = takeH((size_t)NH * HD * DCKV);
    _Float16* wuv16  = takeH((size_t)NH * HD * DCKV);
    _Float16* wkr16  = takeH((size_t)HDR * DIM);
    _Float16* wout16 = takeH((size_t)DIM * DIM);
    _Float16* cq16   = takeH((size_t)MTOK * DCQ);
    _Float16* ckv16  = takeH((size_t)MTOK * DCKV);
    _Float16* q16    = takeH((size_t)MTOK * NH * HDK);
    _Float16* k16    = takeH((size_t)MTOK * NH * HDK);
    _Float16* v16    = takeH((size_t)MTOK * NH * HD);
    _Float16* y16    = takeH((size_t)MTOK * NH * HD);
    float*    cqr    = takeF((size_t)MTOK * NH * HDR);
    float*    krl    = takeF((size_t)MTOK * HDR);

    const dim3 blk(256);
    auto cvt = [&](const float* s, _Float16* d, size_t n) {
        mla_cvt_h<<<(unsigned)((n / 8 + 255) / 256), blk, 0, stream>>>(s, d, (int)n);
    };

    // one-time f32 -> f16 (x + weights)
    cvt(x,     x16,    (size_t)MTOK * DIM);
    cvt(W_dq,  wdq16,  (size_t)DCQ * DIM);
    cvt(W_uq,  wuq16,  (size_t)NH * HD * DCQ);
    cvt(W_qr,  wqr16,  (size_t)NH * HDR * DCQ);
    cvt(W_dkv, wdkv16, (size_t)DCKV * DIM);
    cvt(W_uk,  wuk16,  (size_t)NH * HD * DCKV);
    cvt(W_uv,  wuv16,  (size_t)NH * HD * DCKV);
    cvt(W_kr,  wkr16,  (size_t)HDR * DIM);
    cvt(W_out, wout16, (size_t)DIM * DIM);

    const int MB = MTOK / 64;   // 64 row-blocks
    const int IDSH = 30;        // identity column mapping: c>>30 = 0

    // c_q (f16) = x @ W_dq^T
    mla_gemm_h<<<dim3(DCQ / 64, MB), blk, 0, stream>>>(
        x16, wdq16, b_dq, MTOK, DCQ, DIM, nullptr, 0, cq16, DCQ, IDSH, 0, 0);
    // c_kv (f16) = x @ W_dkv^T
    mla_gemm_h<<<dim3(DCKV / 64, MB), blk, 0, stream>>>(
        x16, wdkv16, b_dkv, MTOK, DCKV, DIM, nullptr, 0, ckv16, DCKV, IDSH, 0, 0);
    // c_q_r (f32) = c_q @ W_qr^T     (rope input)
    mla_gemm_h<<<dim3((NH * HDR) / 64, MB), blk, 0, stream>>>(
        cq16, wqr16, b_qr, MTOK, NH * HDR, DCQ, cqr, NH * HDR, nullptr, 0, IDSH, 0, 0);
    // k_r_lin (f32) = x @ W_kr^T     (rope input)
    mla_gemm_h<<<dim3(HDR / 64, MB), blk, 0, stream>>>(
        x16, wkr16, b_kr, MTOK, HDR, DIM, krl, HDR, nullptr, 0, IDSH, 0, 0);
    // q base -> q16[..., h*192 + 0..127]  (group remap 128 -> 192, shift 7)
    mla_gemm_h<<<dim3((NH * HD) / 64, MB), blk, 0, stream>>>(
        cq16, wuq16, b_uq, MTOK, NH * HD, DCQ, nullptr, 0, q16, NH * HDK, 7, HDK, 0);
    // k base -> k16[..., h*192 + 0..127]
    mla_gemm_h<<<dim3((NH * HD) / 64, MB), blk, 0, stream>>>(
        ckv16, wuk16, b_uk, MTOK, NH * HD, DCKV, nullptr, 0, k16, NH * HDK, 7, HDK, 0);
    // v -> v16 (plain layout)
    mla_gemm_h<<<dim3((NH * HD) / 64, MB), blk, 0, stream>>>(
        ckv16, wuv16, b_uv, MTOK, NH * HD, DCKV, nullptr, 0, v16, NH * HD, IDSH, 0, 0);
    // RoPE into q16/k16 [..., 128..191]
    mla_rope_q<<<(B_ * S_ * NH * HDR) / 256, blk, 0, stream>>>(cqr, fcos, fsin, q16);
    mla_rope_k<<<(B_ * S_ * HDR) / 256, blk, 0, stream>>>(krl, fcos, fsin, k16);
    // causal flash attention -> y16
    mla_attn<<<dim3(S_ / BQ, NH, B_), dim3(128), 0, stream>>>(q16, k16, v16, y16);
    // out (f32) = y @ W_out^T + b_out -> d_out
    mla_gemm_h<<<dim3(DIM / 64, MB), blk, 0, stream>>>(
        y16, wout16, b_out, MTOK, DIM, DIM, (float*)d_out, DIM, nullptr, 0, IDSH, 0, 0);
}
